// ET_71614284694144
// MI455X (gfx1250) — matrix operand order repaired and checked by reading, not compiled, not run on hardware
//
#include <hip/hip_runtime.h>
#include <hip/hip_bf16.h>
#include <math.h>

// ---------------------------------------------------------------------------
// Energy Transformer (ViT) forward on MI455X / gfx1250.
// All heavy GEMMs run on v_wmma_f32_16x16x32_bf16 (wave32, 16x16 tiles).
// ---------------------------------------------------------------------------

typedef __bf16 v16bf __attribute__((ext_vector_type(16)));
typedef float  v8f   __attribute__((ext_vector_type(8)));

union FragU { uint4 u[2]; v16bf v; };

#define BB   32
#define NT   197
#define DD   768
#define HH   12
#define DKK  64
#define HNN  3072
#define TOK  (BB*NT)      // 6304  (divisible by 32)
#define NPATCH 196
#define MPAT (BB*NPATCH)  // 6272
#define NPAD 208          // 197 -> 13*16
#define KPAD 224          // 197 -> 7*32
#define BH   (BB*HH)      // 384
#define BETA 0.125f       // 1/sqrt(64)

// -------------------------- workspace layout -------------------------------
constexpr size_t SZ_STATE = (size_t)TOK*DD*4;        // f32 state
constexpr size_t SZ_TOKD  = (size_t)TOK*DD*2;        // bf16 token x 768
constexpr size_t SZ_QT    = (size_t)BH*DKK*KPAD*2;   // [b,h,e,i] bf16
constexpr size_t SZ_S     = (size_t)BH*NPAD*NPAD*4;  // f32 scores (aliases hid)
constexpr size_t SZ_P     = (size_t)BH*NPAD*KPAD*2;  // bf16 probs (aliases patches)
constexpr size_t SZ_DDw   = (size_t)DD*DD*2;
constexpr size_t SZ_XIw   = (size_t)HNN*DD*2;
constexpr size_t SZ_FCw   = (size_t)1024*DD*2;
constexpr size_t SZ_CLSB  = (size_t)BB*DD*2;

constexpr size_t OFF_STATE = 0;
constexpr size_t OFF_G     = OFF_STATE + SZ_STATE;
constexpr size_t OFF_Q     = OFF_G    + SZ_TOKD;    // also T1
constexpr size_t OFF_K     = OFF_Q    + SZ_TOKD;    // also T2
constexpr size_t OFF_QT    = OFF_K    + SZ_TOKD;
constexpr size_t OFF_KT    = OFF_QT   + SZ_QT;
constexpr size_t OFF_S     = OFF_KT   + SZ_QT;      // also hid (38.7MB < 66.4MB)
constexpr size_t OFF_P     = OFF_S    + SZ_S;       // also patches (9.6MB < 35.8MB)
constexpr size_t OFF_PT    = OFF_P    + SZ_P;
constexpr size_t OFF_WEMB  = OFF_PT   + SZ_P;
constexpr size_t OFF_WQNK  = OFF_WEMB + SZ_DDw;
constexpr size_t OFF_WKNK  = OFF_WQNK + SZ_DDw;
constexpr size_t OFF_WQ2   = OFF_WKNK + SZ_DDw;
constexpr size_t OFF_WK2   = OFF_WQ2  + SZ_DDw;
constexpr size_t OFF_XIT   = OFF_WK2  + SZ_DDw;
constexpr size_t OFF_XI2   = OFF_XIT  + SZ_XIw;
constexpr size_t OFF_FCW   = OFF_XI2  + SZ_XIw;
constexpr size_t OFF_CLSB  = OFF_FCW  + SZ_FCw;

// ------------------------------ WMMA core ----------------------------------
__device__ __forceinline__ v8f wmma_bf16(v16bf a, v16bf b, v8f c) {
    return __builtin_amdgcn_wmma_f32_16x16x32_bf16(false, a, false, b,
                                                   (short)0, c, false, false);
}
// A fragment (16x32, 16-bit): lane m = l&15, kbase = (l>>4)*8,
// elems 0..7 -> K = kbase..kbase+7, elems 8..15 -> K = kbase+16..kbase+23.
// B fragment (32x16, 16-bit, stored [N][K]): lane n = l&15, kbase = (l>>4)*8,
// elems 0..7 -> K = kbase..kbase+7, elems 8..15 -> K = kbase+16..kbase+23.

// ------------------------- generic GEMM (A * B^T) --------------------------
// One wave per 32x64 output tile. A[M][K] bf16 (lda), B[N][K] bf16 (ldb).
enum { EPI_EMBED = 0, EPI_QK, EPI_RELU, EPI_ADD, EPI_FC };

template<int EPI>
__global__ __launch_bounds__(32)
void k_gemm(const __bf16* __restrict__ A, const __bf16* __restrict__ Bm,
            int K, int lda, int ldb, int ldc, int Nvalid,
            float* __restrict__ outf, __bf16* __restrict__ ob0,
            __bf16* __restrict__ ob1,
            const float* __restrict__ aux0, const float* __restrict__ aux1)
{
    const int lane = threadIdx.x;
    const int m0 = blockIdx.x * 32, n0 = blockIdx.y * 64;
    const int amr = m0 + (lane & 15);
    const int ak  = (lane >> 4) << 3;
    const int bnr = n0 + (lane & 15);
    const int bk  = (lane >> 4) << 3;

    const __bf16* ap0 = A + (size_t)amr * lda + ak;
    const __bf16* ap1 = ap0 + (size_t)16 * lda;

    v8f c[2][4] = {};
    for (int k = 0; k < K; k += 32) {
        if (k + 32 < K) __builtin_prefetch(ap0 + k + 32, 0, 0);
        FragU a0, a1;
        a0.u[0] = *(const uint4*)(ap0 + k);
        a0.u[1] = *(const uint4*)(ap0 + k + 16);
        a1.u[0] = *(const uint4*)(ap1 + k);
        a1.u[1] = *(const uint4*)(ap1 + k + 16);
#pragma unroll
        for (int t = 0; t < 4; ++t) {
            const __bf16* bp = Bm + (size_t)(bnr + 16 * t) * ldb + bk + k;
            FragU bb;
            bb.u[0] = *(const uint4*)bp;
            bb.u[1] = *(const uint4*)(bp + 16);
            c[0][t] = wmma_bf16(a0.v, bb.v, c[0][t]);
            c[1][t] = wmma_bf16(a1.v, bb.v, c[1][t]);
        }
    }

    const int nlo = lane & 15, mhi = (lane >> 4) * 8;
#pragma unroll
    for (int i = 0; i < 2; ++i) {
#pragma unroll
        for (int t = 0; t < 4; ++t) {
#pragma unroll
            for (int r = 0; r < 8; ++r) {
                const int m = m0 + 16 * i + mhi + r;
                const int n = n0 + 16 * t + nlo;
                const float v = c[i][t][r];
                if constexpr (EPI == EPI_EMBED) {
                    // m -> (batch, patch); store state row b*197+1+p
                    const int b = m / NPATCH, p = m % NPATCH;
                    outf[((size_t)b * NT + 1 + p) * DD + n] =
                        v + aux0[n] + aux1[(size_t)(1 + p) * DD + n];
                } else if constexpr (EPI == EPI_QK) {
                    const int b = m / NT, ii = m % NT;
                    const int h = n >> 6, e = n & 63;
                    ob0[(size_t)m * DD + n] = (__bf16)v;
                    ob1[(((size_t)b * HH + h) * DKK + e) * KPAD + ii] = (__bf16)v;
                } else if constexpr (EPI == EPI_RELU) {
                    ob0[(size_t)m * ldc + n] = (__bf16)fmaxf(v, 0.0f);
                } else if constexpr (EPI == EPI_ADD) {
                    outf[(size_t)m * ldc + n] += v;
                } else { // EPI_FC
                    if (n < Nvalid)
                        outf[(size_t)m * Nvalid + n] = v + aux0[n];
                }
            }
        }
    }
}

// ----------------------- attention scores  a = Q K^T -----------------------
// grid (13,13,384); one wave per 16x16 tile, K = 64 (2 WMMA steps).
__global__ __launch_bounds__(32)
void k_scores(const __bf16* __restrict__ Q, const __bf16* __restrict__ Kb,
              float* __restrict__ S)
{
    const int bh = blockIdx.z;
    const int b = bh / HH, h = bh % HH;
    const int i0 = blockIdx.x * 16, j0 = blockIdx.y * 16;
    const int lane = threadIdx.x;
    const int am = i0 + (lane & 15), ak = (lane >> 4) << 3;
    const int bn = j0 + (lane & 15), bk = (lane >> 4) << 3;

    v8f c = {};
    for (int k = 0; k < DKK; k += 32) {
        FragU a, bb;
        if (am < NT) {
            const __bf16* p = Q + (size_t)(b * NT + am) * DD + h * DKK + ak + k;
            a.u[0] = *(const uint4*)p; a.u[1] = *(const uint4*)(p + 16);
        } else { a.u[0] = make_uint4(0,0,0,0); a.u[1] = make_uint4(0,0,0,0); }
        if (bn < NT) {
            const __bf16* p = Kb + (size_t)(b * NT + bn) * DD + h * DKK + bk + k;
            bb.u[0] = *(const uint4*)p; bb.u[1] = *(const uint4*)(p + 16);
        } else { bb.u[0] = make_uint4(0,0,0,0); bb.u[1] = make_uint4(0,0,0,0); }
        c = wmma_bf16(a.v, bb.v, c);
    }
    const int nlo = lane & 15, mhi = (lane >> 4) * 8;
#pragma unroll
    for (int r = 0; r < 8; ++r) {
        const int m = i0 + mhi + r, n = j0 + nlo;
        S[((size_t)bh * NPAD + m) * NPAD + n] = c[r];
    }
}

// -------------------- batched  T = P * V   (per b,h head) ------------------
// A = P[bh] (208 x 224, lda 224), B = Vt[bh] ([e][i], 64 x 224, ldb 224).
// grid (13,1,384); one wave computes 16 rows x 64 cols, K = 224.
__global__ __launch_bounds__(32)
void k_termgemm(const __bf16* __restrict__ Pbase, const __bf16* __restrict__ Vt,
                __bf16* __restrict__ Tout)
{
    const int bh = blockIdx.z;
    const int b = bh / HH, h = bh % HH;
    const int i0 = blockIdx.x * 16;
    const __bf16* A  = Pbase + (size_t)bh * NPAD * KPAD;
    const __bf16* Bm = Vt    + (size_t)bh * DKK  * KPAD;
    const int lane = threadIdx.x;
    const int am = i0 + (lane & 15), ak = (lane >> 4) << 3;
    const int bn = lane & 15,        bk = (lane >> 4) << 3;

    v8f c[4] = {};
    for (int k = 0; k < KPAD; k += 32) {
        FragU a;
        const __bf16* ap = A + (size_t)am * KPAD + ak + k;
        a.u[0] = *(const uint4*)ap; a.u[1] = *(const uint4*)(ap + 16);
#pragma unroll
        for (int t = 0; t < 4; ++t) {
            const __bf16* bp = Bm + (size_t)(bn + 16 * t) * KPAD + bk + k;
            FragU bb;
            bb.u[0] = *(const uint4*)bp; bb.u[1] = *(const uint4*)(bp + 16);
            c[t] = wmma_bf16(a.v, bb.v, c[t]);
        }
    }
    const int nlo = lane & 15, mhi = (lane >> 4) * 8;
#pragma unroll
    for (int t = 0; t < 4; ++t)
#pragma unroll
        for (int r = 0; r < 8; ++r) {
            const int i = i0 + mhi + r;
            if (i < NT) {
                const int e = 16 * t + nlo;
                Tout[(size_t)(b * NT + i) * DD + h * DKK + e] = (__bf16)c[t][r];
            }
        }
}

// ------------------------------ softmax rows -------------------------------
__global__ void k_softmax(const float* __restrict__ S, __bf16* __restrict__ P,
                          __bf16* __restrict__ Pt)
{
    const int i = blockIdx.x, bh = blockIdx.y;
    const float* row = S + ((size_t)bh * NPAD + i) * NPAD;
    __shared__ float red[64];
    const int t = threadIdx.x;
    float mx = -3.4e38f;
    for (int j = t; j < NT; j += 64) mx = fmaxf(mx, BETA * row[j]);
    red[t] = mx; __syncthreads();
    for (int s = 32; s > 0; s >>= 1) { if (t < s) red[t] = fmaxf(red[t], red[t + s]); __syncthreads(); }
    mx = red[0]; __syncthreads();
    float sum = 0.0f;
    for (int j = t; j < NT; j += 64) sum += __expf(BETA * row[j] - mx);
    red[t] = sum; __syncthreads();
    for (int s = 32; s > 0; s >>= 1) { if (t < s) red[t] += red[t + s]; __syncthreads(); }
    const float inv = 1.0f / red[0];
    for (int j = t; j < NT; j += 64) {
        const float p = __expf(BETA * row[j] - mx) * inv;
        P [((size_t)bh * NPAD + i) * KPAD + j] = (__bf16)p;
        Pt[((size_t)bh * NPAD + j) * KPAD + i] = (__bf16)p;
    }
}

// ------------------------------ layernorms ---------------------------------
__global__ void k_energy_ln(const float* __restrict__ state, __bf16* __restrict__ g,
                            const float* __restrict__ gamma, const float* __restrict__ delta)
{
    const int row = blockIdx.x, t = threadIdx.x;
    const float* x = state + (size_t)row * DD;
    const float v0 = x[t], v1 = x[t + 256], v2 = x[t + 512];
    __shared__ float s1[256], s2[256];
    s1[t] = v0 + v1 + v2;
    s2[t] = v0 * v0 + v1 * v1 + v2 * v2;
    __syncthreads();
    for (int s = 128; s > 0; s >>= 1) { if (t < s) { s1[t] += s1[t + s]; s2[t] += s2[t + s]; } __syncthreads(); }
    const float mu  = s1[0] * (1.0f / DD);
    const float var = s2[0] * (1.0f / DD) - mu * mu;
    const float rs  = rsqrtf(var + 1e-5f) * gamma[0];
    __bf16* go = g + (size_t)row * DD;
    go[t]       = (__bf16)((v0 - mu) * rs + delta[t]);
    go[t + 256] = (__bf16)((v1 - mu) * rs + delta[t + 256]);
    go[t + 512] = (__bf16)((v2 - mu) * rs + delta[t + 512]);
}

__global__ void k_cls_ln(const float* __restrict__ state, __bf16* __restrict__ clsb,
                         const float* __restrict__ lw, const float* __restrict__ lb)
{
    const int b = blockIdx.x, t = threadIdx.x;
    const float* x = state + (size_t)b * NT * DD;
    const float v0 = x[t], v1 = x[t + 256], v2 = x[t + 512];
    __shared__ float s1[256], s2[256];
    s1[t] = v0 + v1 + v2;
    s2[t] = v0 * v0 + v1 * v1 + v2 * v2;
    __syncthreads();
    for (int s = 128; s > 0; s >>= 1) { if (t < s) { s1[t] += s1[t + s]; s2[t] += s2[t + s]; } __syncthreads(); }
    const float mu  = s1[0] * (1.0f / DD);
    const float var = s2[0] * (1.0f / DD) - mu * mu;
    const float rs  = rsqrtf(var + 1e-5f);
    __bf16* go = clsb + (size_t)b * DD;
    go[t]       = (__bf16)((v0 - mu) * rs * lw[t]       + lb[t]);
    go[t + 256] = (__bf16)((v1 - mu) * rs * lw[t + 256] + lb[t + 256]);
    go[t + 512] = (__bf16)((v2 - mu) * rs * lw[t + 512] + lb[t + 512]);
}

// -------------------------- small utility kernels --------------------------
__global__ void k_cls_init(const float* __restrict__ cls, const float* __restrict__ pos,
                           float* __restrict__ state)
{
    const int b = blockIdx.x;
    for (int j = threadIdx.x; j < DD; j += 256)
        state[(size_t)b * NT * DD + j] = cls[j] + pos[j];
}

__global__ void k_patchify(const float* __restrict__ x, __bf16* __restrict__ ptc)
{
    const int idx = blockIdx.x * 256 + threadIdx.x;
    if (idx >= MPAT * DD) return;
    const int m = idx / DD, f = idx % DD;
    const int b = m / NPATCH, p = m % NPATCH;
    const int py = p / 14, px = p % 14;
    const int c = f % 3, q = f / 3;
    const int iy = q / 16, ix = q % 16;
    const size_t src = (((size_t)b * 3 + c) * 224 + (py * 16 + iy)) * 224 + (px * 16 + ix);
    ptc[idx] = (__bf16)x[src];
}

__global__ void k_zero16(__bf16* __restrict__ p, int n)
{ int i = blockIdx.x * 256 + threadIdx.x; if (i < n) p[i] = (__bf16)0.0f; }

__global__ void k_cvt(const float* __restrict__ s, __bf16* __restrict__ d, int n)
{ int i = blockIdx.x * 256 + threadIdx.x; if (i < n) d[i] = (__bf16)s[i]; }

// Wnk[col=h*64+e][d] = w[h,d,e]
__global__ void k_cvt_wnk(const float* __restrict__ w, __bf16* __restrict__ d)
{
    const int i = blockIdx.x * 256 + threadIdx.x;
    if (i >= DD * DD) return;
    const int col = i / DD, dd = i % DD;
    const int h = col >> 6, e = col & 63;
    d[i] = (__bf16)w[((size_t)h * DD + dd) * DKK + e];
}
// W2[d][he] = w[h,d,e]
__global__ void k_cvt_w2(const float* __restrict__ w, __bf16* __restrict__ d)
{
    const int i = blockIdx.x * 256 + threadIdx.x;
    if (i >= DD * DD) return;
    const int dd = i / DD, he = i % DD;
    const int h = he >> 6, e = he & 63;
    d[i] = (__bf16)w[((size_t)h * DD + dd) * DKK + e];
}
// XiT[m][d] = xi[d][m]
__global__ void k_cvt_xiT(const float* __restrict__ xi, __bf16* __restrict__ d)
{
    const int i = blockIdx.x * 256 + threadIdx.x;
    if (i >= HNN * DD) return;
    const int m = i / DD, dd = i % DD;
    d[i] = (__bf16)xi[(size_t)dd * HNN + m];
}
// fc rows padded 1000 -> 1024 with zeros
__global__ void k_cvt_fc(const float* __restrict__ w, __bf16* __restrict__ d)
{
    const int i = blockIdx.x * 256 + threadIdx.x;
    if (i >= 1024 * DD) return;
    const int n = i / DD, kk = i % DD;
    d[i] = (n < 1000) ? (__bf16)w[(size_t)n * DD + kk] : (__bf16)0.0f;
}

// ------------------------------- launcher ----------------------------------
extern "C" void kernel_launch(void* const* d_in, const int* in_sizes, int n_in,
                              void* d_out, int out_size, void* d_ws, size_t ws_size,
                              hipStream_t stream)
{
    (void)in_sizes; (void)n_in; (void)out_size; (void)ws_size;
    const float* x        = (const float*)d_in[0];
    const float* emb_w    = (const float*)d_in[1];
    const float* emb_b    = (const float*)d_in[2];
    const float* cls_tok  = (const float*)d_in[3];
    const float* pos_emb  = (const float*)d_in[4];
    const float* gamma    = (const float*)d_in[5];
    const float* delta    = (const float*)d_in[6];
    const float* wq       = (const float*)d_in[7];
    const float* wk       = (const float*)d_in[8];
    const float* xi       = (const float*)d_in[9];
    const float* ln_w     = (const float*)d_in[10];
    const float* ln_b     = (const float*)d_in[11];
    const float* fc_w     = (const float*)d_in[12];
    const float* fc_b     = (const float*)d_in[13];
    float* out = (float*)d_out;

    char* ws = (char*)d_ws;
    float*  state   = (float*)(ws + OFF_STATE);
    __bf16* g       = (__bf16*)(ws + OFF_G);
    __bf16* Qbf     = (__bf16*)(ws + OFF_Q);   __bf16* T1 = Qbf;   // aliased
    __bf16* Kbf     = (__bf16*)(ws + OFF_K);   __bf16* T2 = Kbf;   // aliased
    __bf16* Qt      = (__bf16*)(ws + OFF_QT);
    __bf16* Kt      = (__bf16*)(ws + OFF_KT);
    float*  S       = (float*)(ws + OFF_S);
    __bf16* hid     = (__bf16*)(ws + OFF_S);                       // aliased
    __bf16* P       = (__bf16*)(ws + OFF_P);
    __bf16* patches = (__bf16*)(ws + OFF_P);                       // aliased
    __bf16* Pt      = (__bf16*)(ws + OFF_PT);
    __bf16* w_emb   = (__bf16*)(ws + OFF_WEMB);
    __bf16* w_qnk   = (__bf16*)(ws + OFF_WQNK);
    __bf16* w_knk   = (__bf16*)(ws + OFF_WKNK);
    __bf16* w_q2    = (__bf16*)(ws + OFF_WQ2);
    __bf16* w_k2    = (__bf16*)(ws + OFF_WK2);
    __bf16* w_xit   = (__bf16*)(ws + OFF_XIT);
    __bf16* w_xi2   = (__bf16*)(ws + OFF_XI2);
    __bf16* w_fc    = (__bf16*)(ws + OFF_FCW);
    __bf16* clsb    = (__bf16*)(ws + OFF_CLSB);

    auto blks = [](int n) { return (n + 255) / 256; };

    // ---- weight repacks (f32 -> bf16, [N][K] layouts) ----
    k_cvt     <<<blks(DD*DD), 256, 0, stream>>>(emb_w, w_emb, DD*DD);
    k_cvt_wnk <<<blks(DD*DD), 256, 0, stream>>>(wq, w_qnk);
    k_cvt_wnk <<<blks(DD*DD), 256, 0, stream>>>(wk, w_knk);
    k_cvt_w2  <<<blks(DD*DD), 256, 0, stream>>>(wq, w_q2);
    k_cvt_w2  <<<blks(DD*DD), 256, 0, stream>>>(wk, w_k2);
    k_cvt_xiT <<<blks(HNN*DD), 256, 0, stream>>>(xi, w_xit);
    k_cvt     <<<blks(HNN*DD), 256, 0, stream>>>(xi, w_xi2, HNN*DD);
    k_cvt_fc  <<<blks(1024*DD), 256, 0, stream>>>(fc_w, w_fc);

    // ---- patchify + embed GEMM + cls/pos ----
    k_patchify<<<blks(MPAT*DD), 256, 0, stream>>>(x, patches);
    k_cls_init<<<BB, 256, 0, stream>>>(cls_tok, pos_emb, state);
    k_gemm<EPI_EMBED><<<dim3(MPAT/32, DD/64), 32, 0, stream>>>(
        patches, w_emb, DD, DD, DD, DD, DD, state, nullptr, nullptr, emb_b, pos_emb);

    // ---- zero the padded regions of Qt/Kt/P/Pt (pads stay zero all iters) ----
    k_zero16<<<blks(BH*DKK*KPAD), 256, 0, stream>>>(Qt, BH*DKK*KPAD);
    k_zero16<<<blks(BH*DKK*KPAD), 256, 0, stream>>>(Kt, BH*DKK*KPAD);
    k_zero16<<<blks(BH*NPAD*KPAD), 256, 0, stream>>>(P,  BH*NPAD*KPAD);
    k_zero16<<<blks(BH*NPAD*KPAD), 256, 0, stream>>>(Pt, BH*NPAD*KPAD);

    // ---- 7 energy-descent iterations ----
    for (int it = 0; it < 7; ++it) {
        k_energy_ln<<<TOK, 256, 0, stream>>>(state, g, gamma, delta);
        // Q = g Wq , K = g Wk (dual layout: token-major + [b,h,e,i])
        k_gemm<EPI_QK><<<dim3(TOK/32, DD/64), 32, 0, stream>>>(
            g, w_qnk, DD, DD, DD, DD, DD, nullptr, Qbf, Qt, nullptr, nullptr);
        k_gemm<EPI_QK><<<dim3(TOK/32, DD/64), 32, 0, stream>>>(
            g, w_knk, DD, DD, DD, DD, DD, nullptr, Kbf, Kt, nullptr, nullptr);
        // a = Q K^T ; P = softmax(beta a)
        k_scores <<<dim3(13, 13, BH), 32, 0, stream>>>(Qbf, Kbf, S);
        k_softmax<<<dim3(NT, BH), 64, 0, stream>>>(S, P, Pt);
        // T1 = P K ,  T2 = P^T Q   (per head)
        k_termgemm<<<dim3(13, 1, BH), 32, 0, stream>>>(P,  Kt, T1);
        k_termgemm<<<dim3(13, 1, BH), 32, 0, stream>>>(Pt, Qt, T2);
        // state += T1 Wq^T + T2 Wk^T   (attention gradient, sign folded)
        k_gemm<EPI_ADD><<<dim3(TOK/32, DD/64), 32, 0, stream>>>(
            T1, w_q2, DD, DD, DD, DD, DD, state, nullptr, nullptr, nullptr, nullptr);
        k_gemm<EPI_ADD><<<dim3(TOK/32, DD/64), 32, 0, stream>>>(
            T2, w_k2, DD, DD, DD, DD, DD, state, nullptr, nullptr, nullptr, nullptr);
        // hid = relu(g Xi) ; state += hid Xi^T   (Hopfield gradient)
        k_gemm<EPI_RELU><<<dim3(TOK/32, HNN/64), 32, 0, stream>>>(
            g, w_xit, DD, DD, DD, HNN, HNN, nullptr, hid, nullptr, nullptr, nullptr);
        k_gemm<EPI_ADD><<<dim3(TOK/32, DD/64), 32, 0, stream>>>(
            hid, w_xi2, HNN, HNN, HNN, DD, DD, state, nullptr, nullptr, nullptr, nullptr);
    }

    // ---- final LN on cls token + classifier ----
    k_cls_ln<<<BB, 256, 0, stream>>>(state, clsb, ln_w, ln_b);
    k_gemm<EPI_FC><<<dim3(BB/32, 1024/64), 32, 0, stream>>>(
        clsb, w_fc, DD, DD, DD, 1000, 1000, out, nullptr, nullptr, fc_b, nullptr);
}
